// modi_cgcnn_angle_46248207843562
// MI455X (gfx1250) — matrix operand order repaired and checked by reading, not compiled, not run on hardware
//
#include <hip/hip_runtime.h>
#include <hip/hip_bf16.h>
#include <math.h>

#define NBRF 64
#define ANGF 64
#define NANG 1000000
#define NCRY 2048
static __device__ __host__ constexpr float EPS_ = 1e-5f;

typedef __attribute__((ext_vector_type(16))) _Float16 v16h;
typedef __attribute__((ext_vector_type(8)))  _Float16 v8h;
typedef __attribute__((ext_vector_type(4)))  _Float16 v4h;
typedef __attribute__((ext_vector_type(8)))  float    v8f;

__device__ __forceinline__ v8f wmma_f16(v16h a, v16h b, v8f c) {
  // D = A(16x32 f16) * B(32x16 f16) + C(16x16 f32)
  return __builtin_amdgcn_wmma_f32_16x16x32_f16(false, a, false, b, (short)0, c, false, false);
}

// ---------------------------------------------------------------------------
// K0: zero the segment-statistics region of the workspace
// ---------------------------------------------------------------------------
__global__ void k_zero(float* __restrict__ p, int n) {
  int i = blockIdx.x * 256 + threadIdx.x;
  if (i < n) p[i] = 0.0f;
}

// ---------------------------------------------------------------------------
// K1: fused gather + GEMM  gated = [angle | edge[i0] | edge[i1]] @ W_full.T
//     + per-crystal sum / sumsq / count accumulation (for crystal_norm #1)
// Block: 256 thr (8 waves) -> 64 rows. K=192 in two 96-wide LDS chunks.
// Staging loads are float4 -> v4h (ds_store_b64); output tile is transposed
// through LDS (reusing sW) and stored as coalesced 16B v8h chunks.
// ---------------------------------------------------------------------------
__global__ void __launch_bounds__(256) k_gemm1(
    const float* __restrict__ edge, const float* __restrict__ angle,
    const int* __restrict__ nbr, const int* __restrict__ seg,
    const float* __restrict__ Wfull,
    float* __restrict__ cnt1, float* __restrict__ sum1, float* __restrict__ sq1,
    _Float16* __restrict__ gated)
{
  __shared__ _Float16 sA[64 * 96];    // gathered input rows (f16)
  __shared__ _Float16 sW[128 * 96];   // W_full chunk [n][k] (f16); reused as D tile
  const int tid = threadIdx.x;
  const int w = tid >> 5, l = tid & 31;
  const int h = l >> 4, ln = l & 15;
  const int base = blockIdx.x * 64;
  const int mtile  = w & 3;          // 4 row-tiles of 16
  const int ntbase = (w >> 2) * 4;   // waves 0-3: ntiles 0-3, waves 4-7: 4-7

  v8f C[4] = {};

  for (int c = 0; c < 2; ++c) {
    const int kb = c * 96;
    __syncthreads();
    // stage W chunk: float4 loads -> 4-half LDS stores
    for (int q = tid; q < 128 * 24; q += 256) {
      int n = q / 24, kq = (q - n * 24) * 4;
      float4 v = *(const float4*)(Wfull + n * 192 + kb + kq);
      *(v4h*)&sW[n * 96 + kq] =
          (v4h){(_Float16)v.x, (_Float16)v.y, (_Float16)v.z, (_Float16)v.w};
    }
    // stage gathered A rows: quads never straddle the 64/128 source boundaries
    for (int q = tid; q < 64 * 24; q += 256) {
      int rl = q / 24, kq = (q - rl * 24) * 4;
      int r = base + rl, kg = kb + kq;
      const float* src;
      if (kg < 64)       src = angle + (size_t)r * 64 + kg;
      else if (kg < 128) src = edge + (size_t)nbr[2 * r]     * 64 + (kg - 64);
      else               src = edge + (size_t)nbr[2 * r + 1] * 64 + (kg - 128);
      float4 v = *(const float4*)src;
      *(v4h*)&sA[rl * 96 + kq] =
          (v4h){(_Float16)v.x, (_Float16)v.y, (_Float16)v.z, (_Float16)v.w};
    }
    __syncthreads();

    for (int ks = 0; ks < 3; ++ks) {
      v16h a;
      {
        // A layout: lane(0-15)=row, K slots 0..15 (+16 for upper half)
        const _Float16* ap = &sA[(mtile * 16 + ln) * 96 + ks * 32 + 16 * h];
        #pragma unroll
        for (int j = 0; j < 16; ++j) a[j] = ap[j];
      }
      #pragma unroll
      for (int nt = 0; nt < 4; ++nt) {
        v16h b;
        // B layout: lane(0-15)=col, lanes 16-31 carry K+16; B[k][n] = W[n][k]
        const _Float16* bp = &sW[((ntbase + nt) * 16 + ln) * 96 + ks * 32 + 16 * h];
        #pragma unroll
        for (int j = 0; j < 16; ++j) b[j] = bp[j];
        C[nt] = wmma_f16(a, b, C[nt]);
      }
    }
  }

  // per-crystal statistics (var = E[x^2] - E[x]^2), straight from registers
  #pragma unroll
  for (int nt = 0; nt < 4; ++nt) {
    const int n = (ntbase + nt) * 16 + ln;
    #pragma unroll
    for (int t = 0; t < 8; ++t) {
      const int r = base + mtile * 16 + t + 8 * h;
      float v = C[nt][t];
      const int s = seg[r];
      atomicAdd(&sum1[s * 128 + n], v);
      atomicAdd(&sq1[s * 128 + n], v * v);
    }
  }
  if (ntbase == 0 && ln == 0) {   // one count per row
    #pragma unroll
    for (int t = 0; t < 8; ++t) {
      const int r = base + mtile * 16 + t + 8 * h;
      atomicAdd(&cnt1[seg[r]], 1.0f);
    }
  }

  // transpose D tiles through LDS (sW is dead) and store gated coalesced
  __syncthreads();
  _Float16* sOut = sW;              // 64 rows x 128 cols f16 = 16KB
  #pragma unroll
  for (int nt = 0; nt < 4; ++nt) {
    const int n = (ntbase + nt) * 16 + ln;
    #pragma unroll
    for (int t = 0; t < 8; ++t)
      sOut[(mtile * 16 + t + 8 * h) * 128 + n] = (_Float16)C[nt][t];
  }
  __syncthreads();
  for (int q = tid; q < 64 * 16; q += 256) {     // 1024 chunks of 8 halves (16B)
    int rl = q >> 4, cq = (q & 15) * 8;
    v8h val = *(const v8h*)&sOut[rl * 128 + cq];
    *(v8h*)&gated[(size_t)(base + rl) * 128 + cq] = val;
  }
}

// ---------------------------------------------------------------------------
// K2/K4: finalize stats: sum->mean, sq->inv_std (in place)
// ---------------------------------------------------------------------------
__global__ void k_finalize(const float* __restrict__ cnt,
                           float* __restrict__ sum, float* __restrict__ sq, int F)
{
  int idx = blockIdx.x * 256 + threadIdx.x;
  if (idx >= NCRY * F) return;
  int s = idx / F;
  float c = fmaxf(cnt[s], 1.0f);
  float mean = sum[idx] / c;
  float var = sq[idx] / c - mean * mean;
  sum[idx] = mean;
  sq[idx]  = rsqrtf(fmaxf(var, 0.0f) + EPS_);
}

// ---------------------------------------------------------------------------
// K3: apply norm1, relu(core), tanh(filt @ W_mask.T), summed = filt*core,
//     accumulate stats for crystal_norm #2.  One wave per row.
// ---------------------------------------------------------------------------
__global__ void __launch_bounds__(256) k_gate(
    const _Float16* __restrict__ gated, const int* __restrict__ seg,
    const float* __restrict__ mean1, const float* __restrict__ istd1,
    const float* __restrict__ g1, const float* __restrict__ b1,
    const float* __restrict__ Wmask,
    _Float16* __restrict__ summed,
    float* __restrict__ sum2, float* __restrict__ sq2)
{
  const int w = threadIdx.x >> 5, l = threadIdx.x & 31;
  const int r = blockIdx.x * 8 + w;
  const int s = seg[r];
  const size_t gb = (size_t)r * 128;
  auto nrm = [&](int col) -> float {
    float x = (float)gated[gb + col];
    return (x - mean1[s * 128 + col]) * istd1[s * 128 + col] * g1[col] + b1[col];
  };
  float c0 = fmaxf(nrm(l), 0.0f);
  float c1 = fmaxf(nrm(32 + l), 0.0f);
  float p = nrm(64 + l) * Wmask[l] + nrm(96 + l) * Wmask[32 + l];
  #pragma unroll
  for (int off = 16; off > 0; off >>= 1) p += __shfl_xor(p, off, 32);
  float f = tanhf(p);
  float s0 = f * c0, s1 = f * c1;
  summed[(size_t)r * 64 + l]      = (_Float16)s0;
  summed[(size_t)r * 64 + 32 + l] = (_Float16)s1;
  atomicAdd(&sum2[s * 64 + l], s0);          atomicAdd(&sq2[s * 64 + l], s0 * s0);
  atomicAdd(&sum2[s * 64 + 32 + l], s1);     atomicAdd(&sq2[s * 64 + 32 + l], s1 * s1);
}

// ---------------------------------------------------------------------------
// K5: apply norm2, two residual MLP layers (WMMA 16x64x32, 16x32x64),
//     out = relu(angle + s) / sqrt(2).  One wave per 16-row tile.
// ---------------------------------------------------------------------------
__global__ void __launch_bounds__(128) k_res(
    const _Float16* __restrict__ summed, const int* __restrict__ seg,
    const float* __restrict__ mean2, const float* __restrict__ istd2,
    const float* __restrict__ g2, const float* __restrict__ b2,
    const float* __restrict__ W1a, const float* __restrict__ b1a,
    const float* __restrict__ W1b, const float* __restrict__ b1b,
    const float* __restrict__ W2a, const float* __restrict__ b2a,
    const float* __restrict__ W2b, const float* __restrict__ b2b,
    const float* __restrict__ angle, float* __restrict__ out)
{
  __shared__ float    sS[4][16 * 64];   // per-wave f32 activation tile
  __shared__ _Float16 sH[4][16 * 32];   // per-wave f16 hidden tile
  __shared__ _Float16 sB1a[32 * 64], sB1b[64 * 32], sB2a[32 * 64], sB2b[64 * 32];

  const int tid = threadIdx.x;
  for (int i = tid; i < 2048; i += 128) {
    sB1a[i] = (_Float16)W1a[i]; sB1b[i] = (_Float16)W1b[i];
    sB2a[i] = (_Float16)W2a[i]; sB2b[i] = (_Float16)W2b[i];
  }
  __syncthreads();

  const int w = tid >> 5, l = tid & 31, h = l >> 4, ln = l & 15;
  const int rbase = blockIdx.x * 64 + w * 16;
  float*    S = sS[w];
  _Float16* H = sH[w];

  // load + crystal_norm #2 into the tile
  for (int i = l; i < 16 * 64; i += 32) {
    int rl = i >> 6, c = i & 63;
    int r = rbase + rl;
    int s = seg[r];
    float x = (float)summed[(size_t)r * 64 + c];
    S[i] = (x - mean2[s * 64 + c]) * istd2[s * 64 + c] * g2[c] + b2[c];
  }

  #pragma unroll 1
  for (int layer = 0; layer < 2; ++layer) {
    const _Float16* Ba = layer ? sB2a : sB1a;
    const _Float16* Bb = layer ? sB2b : sB1b;
    const float* ba = layer ? b2a : b1a;
    const float* bb = layer ? b2b : b1b;

    // h = relu(s @ Wa.T + ba): M=16, K=64, N=32
    v8f C[2] = {};
    for (int ks = 0; ks < 2; ++ks) {
      v16h a;
      const float* ap = &S[ln * 64 + ks * 32 + 16 * h];
      #pragma unroll
      for (int j = 0; j < 16; ++j) a[j] = (_Float16)ap[j];
      #pragma unroll
      for (int nt = 0; nt < 2; ++nt) {
        v16h b;
        const _Float16* bp = &Ba[(nt * 16 + ln) * 64 + ks * 32 + 16 * h];
        #pragma unroll
        for (int j = 0; j < 16; ++j) b[j] = bp[j];
        C[nt] = wmma_f16(a, b, C[nt]);
      }
    }
    #pragma unroll
    for (int nt = 0; nt < 2; ++nt) {
      int n = nt * 16 + ln;
      float bias = ba[n];
      #pragma unroll
      for (int t = 0; t < 8; ++t)
        H[(t + 8 * h) * 32 + n] = (_Float16)fmaxf(C[nt][t] + bias, 0.0f);
    }

    // s += h @ Wb.T + bb: M=16, K=32, N=64
    v16h a2;
    {
      const _Float16* ap = &H[ln * 32 + 16 * h];
      #pragma unroll
      for (int j = 0; j < 16; ++j) a2[j] = ap[j];
    }
    v8f D[4] = {};
    #pragma unroll
    for (int nt = 0; nt < 4; ++nt) {
      v16h b;
      const _Float16* bp = &Bb[(nt * 16 + ln) * 32 + 16 * h];
      #pragma unroll
      for (int j = 0; j < 16; ++j) b[j] = bp[j];
      D[nt] = wmma_f16(a2, b, D[nt]);
    }
    #pragma unroll
    for (int nt = 0; nt < 4; ++nt) {
      int n = nt * 16 + ln;
      float bias = bb[n];
      #pragma unroll
      for (int t = 0; t < 8; ++t)
        S[(t + 8 * h) * 64 + n] += D[nt][t] + bias;
    }
  }

  // epilogue: out = relu(angle + s) / sqrt(2)
  const float inv_sqrt2 = 0.70710678118654752440f;
  for (int i = l; i < 16 * 64; i += 32) {
    int rl = i >> 6, c = i & 63;
    int r = rbase + rl;
    float v = angle[(size_t)r * 64 + c] + S[i];
    out[(size_t)r * 64 + c] = inv_sqrt2 * fmaxf(v, 0.0f);
  }
}

// ---------------------------------------------------------------------------
extern "C" void kernel_launch(void* const* d_in, const int* in_sizes, int n_in,
                              void* d_out, int out_size, void* d_ws, size_t ws_size,
                              hipStream_t stream)
{
  (void)in_sizes; (void)n_in; (void)out_size; (void)ws_size;
  const float* edge  = (const float*)d_in[0];
  const float* angle = (const float*)d_in[1];
  const int*   nbr   = (const int*)d_in[2];
  const int*   seg   = (const int*)d_in[3];
  const float* Wfull = (const float*)d_in[4];
  const float* Wmask = (const float*)d_in[5];
  const float* g1  = (const float*)d_in[6];
  const float* b1  = (const float*)d_in[7];
  const float* g2  = (const float*)d_in[8];
  const float* b2  = (const float*)d_in[9];
  const float* W1a = (const float*)d_in[10];
  const float* b1a = (const float*)d_in[11];
  const float* W1b = (const float*)d_in[12];
  const float* b1b = (const float*)d_in[13];
  const float* W2a = (const float*)d_in[14];
  const float* b2a = (const float*)d_in[15];
  const float* W2b = (const float*)d_in[16];
  const float* b2b = (const float*)d_in[17];
  float* out = (float*)d_out;

  // workspace layout
  float* stats = (float*)d_ws;
  float* cnt1 = stats;                      // 2048
  float* sum1 = cnt1 + NCRY;                // 2048*128  -> mean1 after finalize
  float* sq1  = sum1 + NCRY * 128;          // 2048*128  -> istd1 after finalize
  float* sum2 = sq1  + NCRY * 128;          // 2048*64   -> mean2
  float* sq2  = sum2 + NCRY * 64;           // 2048*64   -> istd2
  const size_t statsF = (size_t)NCRY * (1 + 2 * 128 + 2 * 64);
  _Float16* gated  = (_Float16*)((char*)d_ws + ((statsF * 4 + 255) / 256) * 256);
  _Float16* summed = gated + (size_t)NANG * 128;

  k_zero<<<(int)((statsF + 255) / 256), 256, 0, stream>>>(stats, (int)statsF);
  k_gemm1<<<NANG / 64, 256, 0, stream>>>(edge, angle, nbr, seg, Wfull,
                                         cnt1, sum1, sq1, gated);
  k_finalize<<<(NCRY * 128 + 255) / 256, 256, 0, stream>>>(cnt1, sum1, sq1, 128);
  k_gate<<<NANG / 8, 256, 0, stream>>>(gated, seg, sum1, sq1, g1, b1, Wmask,
                                       summed, sum2, sq2);
  k_finalize<<<(NCRY * 64 + 255) / 256, 256, 0, stream>>>(cnt1, sum2, sq2, 64);
  k_res<<<NANG / 64, 128, 0, stream>>>(summed, seg, sum2, sq2, g2, b2,
                                       W1a, b1a, W1b, b1b, W2a, b2a, W2b, b2b,
                                       angle, out);
}
